// Transformer_80453327388692
// MI455X (gfx1250) — matrix operand order repaired and checked by reading, not compiled, not run on hardware
//
#include <hip/hip_runtime.h>
#include <hip/hip_bf16.h>

typedef __attribute__((ext_vector_type(16))) _Float16 v16h;
typedef __attribute__((ext_vector_type(8)))  float    v8f;

#define T_SEQ  1024
#define BATCH  8
#define INDIM  80
#define DMODEL 256
#define NHEAD  4
#define DKDIM  64
#define FFDIM  2048
#define NLAYER 6
#define PE_LEN (2 * T_SEQ - 1)

// ---------------------------------------------------------------------------
// WMMA fragment loaders (f32 memory -> f16 fragments, contiguous k = 16h + s in both A and B)
// A-matrix 16x32 f16: lane m=lane&15, half=lane>>4; elements 0..15 = K[k+16h .. +15].
// B-matrix 32x16 f16: lane n=lane&15, half=lane>>4; elements 0..15 = K[k+16h .. +15].
// ---------------------------------------------------------------------------
__device__ __forceinline__ v16h load_fragA(const float* __restrict__ row, int k, int half) {
    const float* p0 = row + k + 16 * half;
    const float* p1 = p0 + 8;
    v16h f;
#pragma unroll
    for (int i = 0; i < 8; ++i) f[i] = (_Float16)p0[i];
#pragma unroll
    for (int i = 0; i < 8; ++i) f[8 + i] = (_Float16)p1[i];
    return f;
}

__device__ __forceinline__ v16h load_fragA_bias(const float* __restrict__ row,
                                                const float* __restrict__ bias,
                                                int k, int half) {
    int k0 = k + 16 * half;
    v16h f;
#pragma unroll
    for (int i = 0; i < 8; ++i) f[i] = (_Float16)(row[k0 + i] + bias[k0 + i]);
#pragma unroll
    for (int i = 0; i < 8; ++i) f[8 + i] = (_Float16)(row[k0 + 8 + i] + bias[k0 + 8 + i]);
    return f;
}

__device__ __forceinline__ v16h load_fragB(const float* __restrict__ row, int k, int half) {
    const float* p = row + k + 16 * half;
    v16h f;
#pragma unroll
    for (int i = 0; i < 16; ++i) f[i] = (_Float16)p[i];
    return f;
}

__device__ __forceinline__ float fast_rcp(float x) { return __builtin_amdgcn_rcpf(x); }

// ---------------------------------------------------------------------------
// WMMA GEMM: C[M,N] = act( A[M,K] @ W[N,K]^T + bias ) (+ resid)
// One wave computes a 32x64 tile (2 M-subtiles x 4 N-subtiles): per K-step,
// 2 A-fragments + 4 B-fragments feed 8 WMMAs (fragment reuse 4x / 2x).
// 4 waves per block. f32 accumulate. act: 0 = none, 1 = DoubleSwish.
// Requires N % 64 == 0; M-tail handled by clamp + guarded stores.
// ---------------------------------------------------------------------------
__global__ __launch_bounds__(128)
void gemm_wmma(const float* __restrict__ A, const float* __restrict__ W,
               const float* __restrict__ bias, const float* __restrict__ resid,
               float* __restrict__ C, int M, int N, int K, int act) {
    int wave = threadIdx.x >> 5;
    int lane = threadIdx.x & 31;
    int tiles_n = N >> 6;                 // 64-wide
    int tiles_m = (M + 31) >> 5;          // 32-tall
    int tile = blockIdx.x * 4 + wave;
    if (tile >= tiles_m * tiles_n) return;          // wave-uniform exit
    int tm = tile / tiles_n, tn = tile % tiles_n;
    int half = lane >> 4, lm = lane & 15;

    int r0 = tm * 32 + lm;      if (r0 >= M) r0 = M - 1;
    int r1 = tm * 32 + 16 + lm; if (r1 >= M) r1 = M - 1;
    const float* arow0 = A + (size_t)r0 * K;
    const float* arow1 = A + (size_t)r1 * K;
    const float* wrow[4];
#pragma unroll
    for (int nt = 0; nt < 4; ++nt)
        wrow[nt] = W + (size_t)(tn * 64 + nt * 16 + lm) * K;

    v8f acc[2][4] = {};
    for (int k = 0; k < K; k += 32) {
        v16h a0 = load_fragA(arow0, k, half);
        v16h a1 = load_fragA(arow1, k, half);
#pragma unroll
        for (int nt = 0; nt < 4; ++nt) {
            v16h b = load_fragB(wrow[nt], k, half);
            acc[0][nt] = __builtin_amdgcn_wmma_f32_16x16x32_f16(false, a0, false, b,
                                                (short)0, acc[0][nt], false, false);
            acc[1][nt] = __builtin_amdgcn_wmma_f32_16x16x32_f16(false, a1, false, b,
                                                (short)0, acc[1][nt], false, false);
        }
    }

#pragma unroll
    for (int mt = 0; mt < 2; ++mt) {
#pragma unroll
        for (int nt = 0; nt < 4; ++nt) {
            int col = tn * 64 + nt * 16 + lm;
            float bv = bias ? bias[col] : 0.0f;
#pragma unroll
            for (int r = 0; r < 8; ++r) {
                int row = tm * 32 + mt * 16 + r + 8 * half;   // C layout: M = r + 8*half
                if (row < M) {
                    float v = acc[mt][nt][r] + bv;
                    if (act == 1)                              // DoubleSwish
                        v = v * fast_rcp(1.0f + __expf(1.0f - v));
                    size_t idx = (size_t)row * N + col;
                    if (resid) v += resid[idx];
                    C[idx] = v;
                }
            }
        }
    }
}

// ---------------------------------------------------------------------------
// Embedding GEMM (K=80, tiny) fused with BasicNorm.
// ---------------------------------------------------------------------------
__global__ __launch_bounds__(256)
void embed_norm(const float* __restrict__ xin, const float* __restrict__ W,
                const float* __restrict__ b, float* __restrict__ xout) {
    __shared__ float red[256];
    int tok = blockIdx.x;
    int c = threadIdx.x;
    const float* xr = xin + (size_t)tok * INDIM;
    const float* wr = W + (size_t)c * INDIM;
    float acc = b[c];
    for (int k = 0; k < INDIM; ++k) acc += xr[k] * wr[k];
    red[c] = acc * acc;
    __syncthreads();
    for (int s = 128; s > 0; s >>= 1) {
        if (c < s) red[c] += red[c + s];
        __syncthreads();
    }
    float scale = rsqrtf(red[0] * (1.0f / DMODEL) + 0.25f);
    xout[(size_t)tok * DMODEL + c] = acc * scale;
}

__global__ __launch_bounds__(256)
void basic_norm_k(float* __restrict__ x) {
    __shared__ float red[256];
    int tok = blockIdx.x;
    int c = threadIdx.x;
    float v = x[(size_t)tok * DMODEL + c];
    red[c] = v * v;
    __syncthreads();
    for (int s = 128; s > 0; s >>= 1) {
        if (c < s) red[c] += red[c + s];
        __syncthreads();
    }
    x[(size_t)tok * DMODEL + c] = v * rsqrtf(red[0] * (1.0f / DMODEL) + 0.25f);
}

// Rel-PE table: unified form pe[i, 2j] = sin((T-1-i)*div_j), pe[i, 2j+1] = cos(...)
__global__ void pe_kernel(float* __restrict__ pe) {
    int idx = blockIdx.x * blockDim.x + threadIdx.x;
    if (idx >= PE_LEN * (DMODEL / 2)) return;
    int i = idx / (DMODEL / 2), j = idx % (DMODEL / 2);
    float div = __expf((float)(2 * j) * (-9.210340371976184f / (float)DMODEL));
    float ang = (float)(T_SEQ - 1 - i) * div;
    pe[(size_t)i * DMODEL + 2 * j]     = sinf(ang);
    pe[(size_t)i * DMODEL + 2 * j + 1] = cosf(ang);
}

// ---------------------------------------------------------------------------
// Fused rel-position flash attention. One wave (32 threads) per
// (batch, head, 16-query tile). Online softmax over 32-key blocks.
// rel_shift done with in-wave shuffles on the WMMA C layout.
// ---------------------------------------------------------------------------
__global__ __launch_bounds__(32)
void attention(const float* __restrict__ qkv, const float* __restrict__ pproj,
               const float* __restrict__ bias_u, const float* __restrict__ bias_v,
               const int* __restrict__ x_lens, float* __restrict__ y) {
    __shared__ float pl[16 * 32];                      // P tile staging (C->A relayout)
    int lane = threadIdx.x;
    int half = lane >> 4, lm = lane & 15;
    int bid = blockIdx.x;
    int qt = bid & (T_SEQ / 16 - 1);
    int h  = (bid / (T_SEQ / 16)) & (NHEAD - 1);
    int b  = bid / ((T_SEQ / 16) * NHEAD);
    int i0 = qt * 16;
    int len = x_lens[b];
    const float* bu = bias_u + h * DKDIM;
    const float* bv = bias_v + h * DKDIM;

    // Q fragments (A layout, row m = lm), with u/v biases folded in.
    const float* qrow = qkv + (size_t)(b * T_SEQ + i0 + lm) * (3 * DMODEL) + h * DKDIM;
    v16h qu[2], qv[2];
#pragma unroll
    for (int s = 0; s < 2; ++s) {
        qu[s] = load_fragA_bias(qrow, bu, s * 32, half);
        qv[s] = load_fragA_bias(qrow, bv, s * 32, half);
    }

    float Mrow[8], Srow[8];
#pragma unroll
    for (int r = 0; r < 8; ++r) { Mrow[r] = -3.0e38f; Srow[r] = 0.0f; }
    v8f o[4] = {};

    const float scale = 0.125f;                       // DK^-0.5
    int jb_max = (i0 + 15) >> 5;                       // causal: last 32-key block
    for (int jb = 0; jb <= jb_max; ++jb) {
        v8f st[2];
#pragma unroll
        for (int sc = 0; sc < 2; ++sc) {
            int j0s = jb * 32 + sc * 16;
            // ac = (q+u) @ k^T  (B frag: lane = key, contiguous dims)
            const float* krow = qkv + (size_t)(b * T_SEQ + j0s + lm) * (3 * DMODEL)
                                + DMODEL + h * DKDIM;
            v8f ac = {};
#pragma unroll
            for (int s = 0; s < 2; ++s) {
                v16h kb = load_fragB(krow, s * 32, half);
                ac = __builtin_amdgcn_wmma_f32_16x16x32_f16(false, qu[s], false, kb,
                                                            (short)0, ac, false, false);
            }
            // bd band: E[i][e] = (q_i+v) . p[base+e], e = 0..31
            int base = T_SEQ - 16 - i0 + j0s;
            v8f E[2];
#pragma unroll
            for (int eb = 0; eb < 2; ++eb) {
                const float* prow = pproj + (size_t)(base + eb * 16 + lm) * DMODEL
                                    + h * DKDIM;
                v8f e = {};
#pragma unroll
                for (int s = 0; s < 2; ++s) {
                    v16h pb = load_fragB(prow, s * 32, half);
                    e = __builtin_amdgcn_wmma_f32_16x16x32_f16(false, qv[s], false, pb,
                                                               (short)0, e, false, false);
                }
                E[eb] = e;
            }
            // rel_shift gather: bd[m][n] = E[m][15 - m + n]; same VGPR r, same half.
            v8f s_tile;
#pragma unroll
            for (int r = 0; r < 8; ++r) {
                int m = r + 8 * half;
                int e = 15 - m + lm;                  // 0..30
                int src = (e & 15) | (half << 4);
                float v0 = __shfl(E[0][r], src, 32);
                float v1 = __shfl(E[1][r], src, 32);
                float bd = (e < 16) ? v0 : v1;
                int jg = j0s + lm, ig = i0 + m;
                bool masked = (jg > ig) || (jg >= len);
                s_tile[r] = masked ? -1.0e30f : (ac[r] + bd) * scale;
            }
            st[sc] = s_tile;
        }
        // Online softmax update over the 32 columns of this key block.
#pragma unroll
        for (int r = 0; r < 8; ++r) {
            float mx = fmaxf(st[0][r], st[1][r]);
#pragma unroll
            for (int off = 1; off < 16; off <<= 1) mx = fmaxf(mx, __shfl_xor(mx, off, 32));
            float mnew = fmaxf(Mrow[r], mx);
            float alpha = __expf(Mrow[r] - mnew);
            Mrow[r] = mnew;
            float p0 = __expf(st[0][r] - mnew);
            float p1 = __expf(st[1][r] - mnew);
            st[0][r] = p0; st[1][r] = p1;
            float rs = p0 + p1;
#pragma unroll
            for (int off = 1; off < 16; off <<= 1) rs += __shfl_xor(rs, off, 32);
            Srow[r] = Srow[r] * alpha + rs;
#pragma unroll
            for (int d = 0; d < 4; ++d) o[d][r] *= alpha;
        }
        // Stage P through LDS, rebuild as A fragment (16 rows x 32 keys).
        __syncthreads();
#pragma unroll
        for (int r = 0; r < 8; ++r) {
            int m = r + 8 * half;
            pl[m * 32 + lm]      = st[0][r];
            pl[m * 32 + 16 + lm] = st[1][r];
        }
        __syncthreads();
        v16h pa;
        {
            const float* rowp = pl + lm * 32;
            int k0 = 16 * half;
#pragma unroll
            for (int i = 0; i < 8; ++i) pa[i] = (_Float16)rowp[k0 + i];
#pragma unroll
            for (int i = 0; i < 8; ++i) pa[8 + i] = (_Float16)rowp[k0 + 8 + i];
        }
        // O += P @ V  (V frag: lane = dim, gather across 16 keys)
#pragma unroll
        for (int d = 0; d < 4; ++d) {
            v16h vf;
#pragma unroll
            for (int e = 0; e < 16; ++e) {
                int key = jb * 32 + 16 * half + e;
                vf[e] = (_Float16)qkv[(size_t)(b * T_SEQ + key) * (3 * DMODEL)
                                      + 2 * DMODEL + h * DKDIM + d * 16 + lm];
            }
            o[d] = __builtin_amdgcn_wmma_f32_16x16x32_f16(false, pa, false, vf,
                                                          (short)0, o[d], false, false);
        }
    }
    // Normalize and write attention output (B,T,D) with head offset.
#pragma unroll
    for (int r = 0; r < 8; ++r) {
        float inv = fast_rcp(Srow[r]);
        int row = i0 + r + 8 * half;
#pragma unroll
        for (int d = 0; d < 4; ++d)
            y[(size_t)(b * T_SEQ + row) * DMODEL + h * DKDIM + d * 16 + lm] = o[d][r] * inv;
    }
}

__global__ void finalize(const float* __restrict__ x, const int* __restrict__ lens,
                         float* __restrict__ out, int n) {
    int i = blockIdx.x * blockDim.x + threadIdx.x;
    if (i < n) out[i] = x[i];
    if (i < BATCH) ((int*)out)[n + i] = lens[i];
}

// ---------------------------------------------------------------------------
extern "C" void kernel_launch(void* const* d_in, const int* in_sizes, int n_in,
                              void* d_out, int out_size, void* d_ws, size_t ws_size,
                              hipStream_t stream) {
    (void)in_sizes; (void)n_in; (void)out_size; (void)ws_size;
    const float* xin        = (const float*)d_in[0];
    const int*   x_lens     = (const int*)d_in[1];
    const float* embed_W    = (const float*)d_in[2];
    const float* embed_b    = (const float*)d_in[3];
    const float* in_proj_W  = (const float*)d_in[4];
    const float* in_proj_b  = (const float*)d_in[5];
    const float* pos_W      = (const float*)d_in[6];
    const float* pos_bias_u = (const float*)d_in[7];
    const float* pos_bias_v = (const float*)d_in[8];
    const float* out_W      = (const float*)d_in[9];
    const float* out_b      = (const float*)d_in[10];
    const float* ff1_W      = (const float*)d_in[11];
    const float* ff1_b      = (const float*)d_in[12];
    const float* ff2_W      = (const float*)d_in[13];
    const float* ff2_b      = (const float*)d_in[14];

    float* ws = (float*)d_ws;
    size_t off = 0;
    float* x    = ws + off; off += (size_t)BATCH * T_SEQ * DMODEL;     // 8 MB
    float* y    = ws + off; off += (size_t)BATCH * T_SEQ * DMODEL;     // 8 MB
    float* qkv  = ws + off; off += (size_t)BATCH * T_SEQ * 3 * DMODEL; // 24 MB
    float* hbuf = ws + off; off += (size_t)BATCH * T_SEQ * FFDIM;      // 64 MB
    float* pe   = ws + off; off += (size_t)PE_LEN * DMODEL;            // 2 MB
    float* ppj  = ws + off; off += (size_t)PE_LEN * DMODEL;            // 2 MB

    const int M = BATCH * T_SEQ;
    auto tiles = [](int m, int n) { return (((m + 31) / 32)) * (n / 64); };

    embed_norm<<<M, 256, 0, stream>>>(xin, embed_W, embed_b, x);
    pe_kernel<<<(PE_LEN * (DMODEL / 2) + 255) / 256, 256, 0, stream>>>(pe);

    for (int l = 0; l < NLAYER; ++l) {
        // p = pe @ pos_W^T
        gemm_wmma<<<(tiles(PE_LEN, DMODEL) + 3) / 4, 128, 0, stream>>>(
            pe, pos_W + (size_t)l * DMODEL * DMODEL, nullptr, nullptr, ppj,
            PE_LEN, DMODEL, DMODEL, 0);
        // qkv = x @ in_proj_W^T + b
        gemm_wmma<<<(tiles(M, 3 * DMODEL) + 3) / 4, 128, 0, stream>>>(
            x, in_proj_W + (size_t)l * 3 * DMODEL * DMODEL,
            in_proj_b + (size_t)l * 3 * DMODEL, nullptr, qkv, M, 3 * DMODEL, DMODEL, 0);
        // fused rel-pos flash attention -> y
        attention<<<BATCH * NHEAD * (T_SEQ / 16), 32, 0, stream>>>(
            qkv, ppj, pos_bias_u + (size_t)l * NHEAD * DKDIM,
            pos_bias_v + (size_t)l * NHEAD * DKDIM, x_lens, y);
        // x = x + y @ out_W^T + b   (element-wise in-place residual is race-free)
        gemm_wmma<<<(tiles(M, DMODEL) + 3) / 4, 128, 0, stream>>>(
            y, out_W + (size_t)l * DMODEL * DMODEL, out_b + (size_t)l * DMODEL,
            x, x, M, DMODEL, DMODEL, 0);
        // h = DoubleSwish(x @ ff1_W^T + b)
        gemm_wmma<<<(tiles(M, FFDIM) + 3) / 4, 128, 0, stream>>>(
            x, ff1_W + (size_t)l * FFDIM * DMODEL, ff1_b + (size_t)l * FFDIM,
            nullptr, hbuf, M, FFDIM, DMODEL, 1);
        // x = x + h @ ff2_W^T + b
        gemm_wmma<<<(tiles(M, DMODEL) + 3) / 4, 128, 0, stream>>>(
            hbuf, ff2_W + (size_t)l * DMODEL * FFDIM, ff2_b + (size_t)l * DMODEL,
            x, x, M, DMODEL, FFDIM, 0);
        // x = BasicNorm(x)
        basic_norm_k<<<M, 256, 0, stream>>>(x);
    }
    int n = M * DMODEL;
    finalize<<<(n + 255) / 256, 256, 0, stream>>>(x, x_lens, (float*)d_out, n);
}
